// SelfAttentionBlock_71579924955635
// MI455X (gfx1250) — compile-verified, matrix operand order checked by reading, not run on hardware
//
#include <hip/hip_runtime.h>
#include <hip/hip_bf16.h>
#include <math.h>

typedef __bf16  v16bf __attribute__((ext_vector_type(16)));
typedef __bf16  v8bf  __attribute__((ext_vector_type(8)));
typedef float   v8f   __attribute__((ext_vector_type(8)));

constexpr int BB = 2;       // batch
constexpr int S  = 2048;    // sequence
constexpr int DM = 1024;    // model dim
constexpr int H  = 16;      // heads
constexpr int HD = 64;      // head dim
constexpr int BS = BB * S;  // 4096 rows
constexpr float SCALE = 0.125f;  // 1/sqrt(64)

// K index inside a 16-bit WMMA A/B fragment for element e (0..15) of lane.
__device__ __forceinline__ int frag_k(int e, int lane) {
  return ((e < 8) ? e : e + 8) + ((lane & 16) ? 8 : 0);
}

__device__ __forceinline__ v16bf bfcat(v8bf lo, v8bf hi) {
  return __builtin_shufflevector(lo, hi, 0, 1, 2, 3, 4, 5, 6, 7,
                                 8, 9, 10, 11, 12, 13, 14, 15);
}

// A fragment (16x32, row-major source, row = lane&15) — also used for
// "transposed B" (B(k,n) = src[n*ld + k]). K=0..7 / 16..23 per lane are
// contiguous -> two 16B loads per lane (global_load_b128 or ds_load_b128).
__device__ __forceinline__ v16bf load_a16(const __bf16* p, int ld) {
  int lane = threadIdx.x & 31;
  const __bf16* r = p + (lane & 15) * ld + ((lane & 16) ? 8 : 0);
  return bfcat(*(const v8bf*)r, *(const v8bf*)(r + 16));
}

// B fragment from a pre-swizzled 32x16 tile (512 bf16, lane-linear).
__device__ __forceinline__ v16bf load_bswz(const __bf16* blk) {
  int lane = threadIdx.x & 31;
  const __bf16* r = blk + lane * 16;
  return bfcat(*(const v8bf*)r, *(const v8bf*)(r + 8));
}

__device__ __forceinline__ v8f wmma_bf16(v16bf a, v16bf b, v8f c) {
  return __builtin_amdgcn_wmma_f32_16x16x32_bf16(false, a, false, b,
                                                 (short)0, c, false, false);
}

// ---------------- gfx1250 async copy helpers (ASYNCcnt path) ----------------

// Low 32 bits of a generic pointer to LDS == LDS byte address (aperture rules).
__device__ __forceinline__ unsigned lds_off_u32(const void* p) {
  return (unsigned)(uintptr_t)p;
}

// GVS form: mem = SGPR64 base + 32-bit VGPR offset; 16B/lane -> LDS[ldsoff].
__device__ __forceinline__ void async_copy_b128(unsigned lds_byte_off,
                                                const void* sbase,
                                                unsigned voff_bytes) {
  asm volatile("global_load_async_to_lds_b128 %0, %1, %2"
               :
               : "v"(lds_byte_off), "v"(voff_bytes), "s"(sbase)
               : "memory");
}

__device__ __forceinline__ void wait_async0() {
  asm volatile("s_wait_asynccnt 0x0" ::: "memory");
}

// ---------------- conversion / repack kernels ----------------

__global__ void k_cvt(const float* __restrict__ in, __bf16* __restrict__ out, int n) {
  int i = blockIdx.x * blockDim.x + threadIdx.x;
  int stride = gridDim.x * blockDim.x;
  for (; i < n; i += stride) out[i] = (__bf16)in[i];
}

// W [DM x DM] fp32 -> swizzled bf16 W^T fragment blocks: B(k,n) = W[n][k].
__global__ void k_swz_wT(const float* __restrict__ w, __bf16* __restrict__ dst) {
  int idx = blockIdx.x * blockDim.x + threadIdx.x;  // DM*DM threads exactly
  int e = idx & 15, lane = (idx >> 4) & 31, blk = idx >> 9;
  int kt = blk >> 6, nt = blk & 63;
  int k = kt * 32 + frag_k(e, lane);
  int n = nt * 16 + (lane & 15);
  dst[idx] = (__bf16)w[n * DM + k];
}

// Row-major bf16 [R x 1024] -> swizzled fragment blocks: B(k,n) = src[k][n].
__global__ void k_swz(const __bf16* __restrict__ src, __bf16* __restrict__ dst, int total) {
  int idx = blockIdx.x * blockDim.x + threadIdx.x;
  int stride = gridDim.x * blockDim.x;
  for (; idx < total; idx += stride) {
    int e = idx & 15, lane = (idx >> 4) & 31, blk = idx >> 9;
    int kt = blk >> 6, nt = blk & 63;
    int k = kt * 32 + frag_k(e, lane);
    int n = nt * 16 + (lane & 15);
    dst[idx] = src[(size_t)k * DM + n];
  }
}

// ---------------- GEMM: C[BSxDM] = A[BSxDM] @ Bswz(+bias) ----------------
// 256 threads = 8 waves; block tile 64(M) x 128(N); wave tile 32x32.
// Operand panels double-buffered in LDS via async-to-LDS copies (ASYNCcnt),
// overlapping the next K-step's global traffic with this step's WMMAs.

template <bool FINAL>
__global__ void k_gemm(const __bf16* __restrict__ A, const __bf16* __restrict__ Bswz,
                       __bf16* __restrict__ outb, float* __restrict__ outf,
                       const float* __restrict__ bias) {
  __shared__ __align__(16) __bf16 sA[2][64 * 40];  // 64 rows x 32 (+8 pad) bf16
  __shared__ __align__(16) __bf16 sB[2][8 * 512];  // 8 swizzled 32x16 tiles
  const int tid  = threadIdx.x;
  const int lane = tid & 31;
  const int w    = tid >> 5;
  const int m0l  = (w & 1) * 32;   // local tile row
  const int n0l  = (w >> 1) * 32;  // local tile col
  const int bx = blockIdx.x, by = blockIdx.y;

  // A panel: 64 rows x 64B; one b128 instr covers 8 rows; wave w moves rows w*8..w*8+7.
  const __bf16* Abase = A + (size_t)bx * 64 * DM;
  const int arow = w * 8 + (lane >> 2);
  const unsigned avoff0 = (unsigned)(arow * (DM * 2) + (lane & 3) * 16);
  const unsigned aldso  = (unsigned)(arow * 80 + (lane & 3) * 16);
  // B panel: 8KB contiguous (8 consecutive swizzled blocks); wave w moves 1KB.
  const unsigned bvoff0 = (unsigned)(by * 8192 + w * 1024 + lane * 16);
  const unsigned bldso  = (unsigned)(w * 1024 + lane * 16);

  const unsigned aL[2] = { lds_off_u32(&sA[0][0]) + aldso,
                           lds_off_u32(&sA[1][0]) + aldso };
  const unsigned bL[2] = { lds_off_u32(&sB[0][0]) + bldso,
                           lds_off_u32(&sB[1][0]) + bldso };

  auto stage = [&](int kk, int buf) {
    async_copy_b128(aL[buf], Abase, avoff0 + (unsigned)(kk * 2));
    unsigned bv = bvoff0 + (unsigned)((kk >> 5) * 65536);
    async_copy_b128(bL[buf],       Bswz, bv);
    async_copy_b128(bL[buf] + 512, Bswz, bv + 512);
  };

  stage(0, 0);
  v8f c[2][2] = {};
  for (int kk = 0; kk < DM; kk += 32) {
    const int buf = (kk >> 5) & 1;
    wait_async0();            // this wave's stage-kk data is in LDS
    __syncthreads();          // everyone's stage-kk data is in LDS
    if (kk + 32 < DM) stage(kk + 32, buf ^ 1);  // prefetch next stage
    const __bf16* pa = &sA[buf][0];
    const __bf16* pb = &sB[buf][(w >> 1) * 2 * 512];
    v16bf a0 = load_a16(pa + m0l * 40, 40);
    v16bf a1 = load_a16(pa + (m0l + 16) * 40, 40);
    v16bf b0 = load_bswz(pb);
    v16bf b1 = load_bswz(pb + 512);
    c[0][0] = wmma_bf16(a0, b0, c[0][0]);
    c[0][1] = wmma_bf16(a0, b1, c[0][1]);
    c[1][0] = wmma_bf16(a1, b0, c[1][0]);
    c[1][1] = wmma_bf16(a1, b1, c[1][1]);
  }

  const int m0 = bx * 64 + m0l;
  const int n0 = by * 128 + n0l;
  const int rbase = (lane & 16) ? 8 : 0;
#pragma unroll
  for (int i = 0; i < 2; ++i)
#pragma unroll
    for (int j = 0; j < 2; ++j) {
      int n = n0 + j * 16 + (lane & 15);
      float bv = FINAL ? bias[n] : 0.0f;
#pragma unroll
      for (int r = 0; r < 8; ++r) {
        int m = m0 + i * 16 + rbase + r;
        if (FINAL) outf[(size_t)m * DM + n] = c[i][j][r] + bv;
        else       outb[(size_t)m * DM + n] = (__bf16)c[i][j][r];
      }
    }
}

// ---------------- column softmax stats (softmax over QUERY axis) ----------------
// One wave per (b, h, 16-key-column tile); loop q, online max/sum-exp per column.

__global__ void k_colstats(const __bf16* __restrict__ qp, const __bf16* __restrict__ kp,
                           float* __restrict__ cmax, float* __restrict__ csum) {
  int lane = threadIdx.x & 31;
  int bx = blockIdx.x;            // BB*H*(S/16)
  int kt = (bx & 127) * 16;
  int h  = (bx >> 7) & 15;
  int b  = bx >> 11;
  const __bf16* kbase = kp + ((size_t)(b * S + kt)) * DM + h * HD;
  v16bf bk0 = load_a16(kbase, DM);       // B(k=d, n=key) = kp[key][d]
  v16bf bk1 = load_a16(kbase + 32, DM);
  float mx = -INFINITY, sm = 0.0f;
  for (int q0 = 0; q0 < S; q0 += 16) {
    const __bf16* ap = qp + ((size_t)(b * S + q0)) * DM + h * HD;
    if (q0 + 16 < S)
      __builtin_prefetch(ap + (size_t)16 * DM, 0, 0);
    v16bf a0 = load_a16(ap, DM);
    v16bf a1 = load_a16(ap + 32, DM);
    v8f cc = {};
    cc = wmma_bf16(a0, bk0, cc);
    cc = wmma_bf16(a1, bk1, cc);
    float tmax = -INFINITY;
#pragma unroll
    for (int r = 0; r < 8; ++r) tmax = fmaxf(tmax, cc[r] * SCALE);
    float ts = 0.0f;
#pragma unroll
    for (int r = 0; r < 8; ++r) ts += __expf(cc[r] * SCALE - tmax);
    float nm = fmaxf(mx, tmax);
    sm = sm * __expf(mx - nm) + ts * __expf(tmax - nm);
    mx = nm;
  }
  // combine the two row-halves (lanes l and l^16 hold the same column)
  float om = __shfl_xor(mx, 16, 32);
  float os = __shfl_xor(sm, 16, 32);
  float nm = fmaxf(mx, om);
  float tot = sm * __expf(mx - nm) + os * __expf(om - nm);
  if (lane < 16) {
    size_t o = (size_t)(b * H + h) * S + kt + lane;
    cmax[o] = nm;
    csum[o] = tot;
  }
}

// ---------------- attention x V ----------------
// One wave per (b, h, 16-query tile). Loop keys in steps of 32:
// scores (WMMA) -> exp/normalize with precomputed column stats -> LDS ->
// re-fragment as A -> WMMA with swizzled V. Writes sa (fp32) + bf16 copy.

__global__ void k_attnv(const __bf16* __restrict__ qp, const __bf16* __restrict__ kp,
                        const __bf16* __restrict__ vswz,
                        const float* __restrict__ cmax, const float* __restrict__ csum,
                        float* __restrict__ sa, __bf16* __restrict__ sab) {
  __shared__ __align__(16) __bf16 pl[16][40];
  int lane = threadIdx.x & 31;
  int bx = blockIdx.x;
  int qt = (bx & 127) * 16;
  int h  = (bx >> 7) & 15;
  int b  = bx >> 11;
  int bh = b * H + h;
  const __bf16* qbase = qp + ((size_t)(b * S + qt)) * DM + h * HD;
  v16bf a0 = load_a16(qbase, DM);
  v16bf a1 = load_a16(qbase + 32, DM);
  v8f acc[4] = {};
  int rbase = (lane & 16) ? 8 : 0;
  for (int k0 = 0; k0 < S; k0 += 32) {
    if (k0 + 32 < S)
      __builtin_prefetch(kp + ((size_t)(b * S + k0 + 32)) * DM + h * HD, 0, 0);
#pragma unroll
    for (int t = 0; t < 2; ++t) {
      const __bf16* kbase = kp + ((size_t)(b * S + k0 + t * 16)) * DM + h * HD;
      v16bf bk0 = load_a16(kbase, DM);
      v16bf bk1 = load_a16(kbase + 32, DM);
      v8f cc = {};
      cc = wmma_bf16(a0, bk0, cc);
      cc = wmma_bf16(a1, bk1, cc);
      size_t co = (size_t)bh * S + k0 + t * 16 + (lane & 15);
      float m  = cmax[co];
      float rs = 1.0f / csum[co];
#pragma unroll
      for (int r = 0; r < 8; ++r)
        pl[rbase + r][t * 16 + (lane & 15)] =
            (__bf16)(__expf(cc[r] * SCALE - m) * rs);
    }
    __syncthreads();
    v16bf a2 = load_a16(&pl[0][0], 40);
    const __bf16* vb = vswz + ((size_t)((b * S + k0) >> 5) * 64 + h * 4) * 512;
#pragma unroll
    for (int dt = 0; dt < 4; ++dt) {
      v16bf bv = load_bswz(vb + dt * 512);
      acc[dt] = wmma_bf16(a2, bv, acc[dt]);
    }
    __syncthreads();
  }
  float* sap = sa + ((size_t)bh * S + qt) * HD;
  size_t sbase = (size_t)b * S * DM + (size_t)h * S * HD + (size_t)qt * HD;
#pragma unroll
  for (int dt = 0; dt < 4; ++dt)
#pragma unroll
    for (int r = 0; r < 8; ++r) {
      int qq = rbase + r;
      int d  = dt * 16 + (lane & 15);
      float val = acc[dt][r];
      sap[qq * HD + d] = val;                          // output #2 ([B,H,S,hd])
      sab[sbase + (size_t)qq * HD + d] = (__bf16)val;  // reshape view = same layout
    }
}

// ---------------- host launcher ----------------

extern "C" void kernel_launch(void* const* d_in, const int* in_sizes, int n_in,
                              void* d_out, int out_size, void* d_ws, size_t ws_size,
                              hipStream_t stream) {
  (void)in_sizes; (void)n_in; (void)out_size; (void)ws_size;
  const float* q  = (const float*)d_in[0];
  const float* k  = (const float*)d_in[1];
  const float* v  = (const float*)d_in[2];
  const float* Wq = (const float*)d_in[3];
  const float* Wk = (const float*)d_in[4];
  const float* Wv = (const float*)d_in[5];
  const float* Wo = (const float*)d_in[6];
  const float* bo = (const float*)d_in[7];

  constexpr size_t NE = (size_t)BS * DM;  // 4,194,304 elems
  constexpr size_t NW = (size_t)DM * DM;  // 1,048,576 elems

  __bf16* qb  = (__bf16*)d_ws;        // bf16 inputs
  __bf16* kb  = qb + NE;
  __bf16* vb  = kb + NE;
  __bf16* wqs = vb + NE;              // swizzled W^T
  __bf16* wks = wqs + NW;
  __bf16* wvs = wks + NW;
  __bf16* wos = wvs + NW;
  __bf16* qpj = wos + NW;             // projections (bf16, row-major)
  __bf16* kpj = qpj + NE;
  __bf16* vpj = kpj + NE;
  __bf16* vsz = vpj + NE;             // swizzled V
  __bf16* sab = vsz + NE;             // bf16 sa for output GEMM
  float*  cmax = (float*)(sab + NE);  // per-column softmax stats
  float*  csum = cmax + (size_t)BB * H * S;

  float* out = (float*)d_out;   // [B,S,DM]
  float* sa  = out + NE;        // [B,H,S,HD]

  k_cvt<<<2048, 256, 0, stream>>>(q, qb, (int)NE);
  k_cvt<<<2048, 256, 0, stream>>>(k, kb, (int)NE);
  k_cvt<<<2048, 256, 0, stream>>>(v, vb, (int)NE);
  k_swz_wT<<<NW / 256, 256, 0, stream>>>(Wq, wqs);
  k_swz_wT<<<NW / 256, 256, 0, stream>>>(Wk, wks);
  k_swz_wT<<<NW / 256, 256, 0, stream>>>(Wv, wvs);
  k_swz_wT<<<NW / 256, 256, 0, stream>>>(Wo, wos);

  dim3 gg(BS / 64, DM / 128);
  k_gemm<false><<<gg, 256, 0, stream>>>(qb, wqs, qpj, nullptr, nullptr);
  k_gemm<false><<<gg, 256, 0, stream>>>(kb, wks, kpj, nullptr, nullptr);
  k_gemm<false><<<gg, 256, 0, stream>>>(vb, wvs, vpj, nullptr, nullptr);

  k_swz<<<2048, 256, 0, stream>>>(vpj, vsz, (int)NE);

  int gattn = BB * H * (S / 16);  // 4096 single-wave workgroups
  k_colstats<<<gattn, 32, 0, stream>>>(qpj, kpj, cmax, csum);
  k_attnv<<<gattn, 32, 0, stream>>>(qpj, kpj, vsz, cmax, csum, sa, sab);

  k_gemm<true><<<gg, 256, 0, stream>>>(sab, wos, nullptr, out, bo);
}
